// EnhancedBVHRouter_54898271978235
// MI455X (gfx1250) — matrix operand order repaired and checked by reading, not compiled, not run on hardware
//
#include <hip/hip_runtime.h>
#include <hip/hip_bf16.h>

typedef __attribute__((ext_vector_type(16))) __bf16   v16bf;
typedef __attribute__((ext_vector_type(8)))  float    v8f;
typedef __attribute__((ext_vector_type(8)))  unsigned v8u;

__device__ __forceinline__ __bf16 f2bf(float f) {
  unsigned u = __builtin_bit_cast(unsigned, f);
  u = u + 0x7FFFu + ((u >> 16) & 1u);           // round-to-nearest-even
  unsigned short h = (unsigned short)(u >> 16);
  return __builtin_bit_cast(__bf16, h);
}

__device__ __forceinline__ float bf2f(__bf16 h) {
  unsigned u = ((unsigned)__builtin_bit_cast(unsigned short, h)) << 16;
  return __builtin_bit_cast(float, u);
}

__device__ __forceinline__ float gelu_exact(float x) {
  return 0.5f * x * (1.0f + erff(x * 0.70710678118654752f));
}

__device__ __forceinline__ v8f zero8() {
  v8f z = {0.f, 0.f, 0.f, 0.f, 0.f, 0.f, 0.f, 0.f};
  return z;
}

// ---------------------------------------------------------------------------
// Wave-level tile GEMM: acc += A(16xKC) * B(KCx16) via v_wmma_f32_16x16x32_bf16.
//   A  : bf16 row-major in LDS, leading dim lda (elements), rows 16B-aligned.
//   Bt : bf16 column-major weight ([N][ldbt]), pointer pre-offset to the tile's
//        first column and K-chunk base. Rows 32B-aligned.
//   KC : compile-time chunk length, multiple of 32.
// A fragment per lane = two contiguous 8-element (16B) runs -> 2 LDS b128 loads.
// B fragment per lane = 16 contiguous bf16 (32B)            -> 2 vmem b128 loads.
// ---------------------------------------------------------------------------
template <int KC>
__device__ __forceinline__ v8f wmma_tileB(const __bf16* __restrict__ A, int lda,
                                          const __bf16* __restrict__ Bt, int ldbt,
                                          v8f acc) {
  const int lane = threadIdx.x & 31;
  const int half = lane >> 4;                               // 0 or 1
  const __bf16* arow = A  + (lane & 15) * lda  + half * 16; // A: M = lane&15
  const __bf16* brow = Bt + (lane & 15) * ldbt + half * 16; // B: N = lane&15
#pragma unroll 2
  for (int k0 = 0; k0 < KC; k0 += 32) {
    uint4 a0 = *(const uint4*)(arow + k0);        // K = kA0 .. kA0+7
    uint4 a1 = *(const uint4*)(arow + k0 + 8);    // K = kA0+8 .. kA0+15
    v8u au;
    au[0] = a0.x; au[1] = a0.y; au[2] = a0.z; au[3] = a0.w;
    au[4] = a1.x; au[5] = a1.y; au[6] = a1.z; au[7] = a1.w;
    v16bf a = __builtin_bit_cast(v16bf, au);
    v16bf b = *(const v16bf*)(brow + k0);
    acc = __builtin_amdgcn_wmma_f32_16x16x32_bf16(false, a, false, b,
                                                  (short)0, acc, false, false);
  }
  return acc;
}

// Store a 16x16 C/D tile (+bias, optional exact GELU) as bf16, row-major.
__device__ __forceinline__ void store_tile_bf(__bf16* __restrict__ out, int ldo, int n0,
                                              const float* __restrict__ bias,
                                              v8f acc, bool do_gelu) {
  const int lane = threadIdx.x & 31;
  const int half = lane >> 4;
  const int nc   = lane & 15;
  float bv = bias[n0 + nc];
#pragma unroll
  for (int j = 0; j < 8; ++j) {                 // C/D layout: VGPR j -> M = j+half*8
    float v = acc[j] + bv;
    if (do_gelu) v = gelu_exact(v);
    out[(j + half * 8) * ldo + n0 + nc] = f2bf(v);
  }
}

// Store a 16x16 C/D tile (+bias) as f32, row-major.
__device__ __forceinline__ void store_tile_f32(float* __restrict__ out, int ldo, int n0,
                                               const float* __restrict__ bias, v8f acc) {
  const int lane = threadIdx.x & 31;
  const int half = lane >> 4;
  const int nc   = lane & 15;
  float bv = bias[n0 + nc];
#pragma unroll
  for (int j = 0; j < 8; ++j)
    out[(j + half * 8) * ldo + n0 + nc] = acc[j] + bv;
}

// One BVH routing level for the block's 16 rows. Activations are bf16 in LDS.
template <int IN_DIM>
__device__ __forceinline__ void run_level(const __bf16* Hin, int ldH,
                                          const float* w3, const float* b3,
                                          const float* centers,
                                          const __bf16* f1t, const float* f1b,
                                          const __bf16* f2t, const float* f2b,
                                          const float* rw, const float* rb,
                                          __bf16* Tmp /*16x128*/, __bf16* Fout /*16x128*/,
                                          float* pos /*16x3*/, float* lg /*16x4*/,
                                          float* probs /*16x4*/) {
  const int tid = threadIdx.x;
  const int wid = tid >> 5;
  { // Tmp = GELU(Hin @ f1w + f1b), N=128: one 16x16 tile per wave
    int n0 = wid * 16;
    v8f acc = wmma_tileB<IN_DIM>(Hin, ldH, f1t + n0 * IN_DIM, IN_DIM, zero8());
    store_tile_bf(Tmp, 128, n0, f1b, acc, true);
  }
  __syncthreads();
  { // Fout = Tmp @ f2w + f2b, K=N=128
    int n0 = wid * 16;
    v8f acc = wmma_tileB<128>(Tmp, 128, f2t + n0 * 128, 128, zero8());
    store_tile_bf(Fout, 128, n0, f2b, acc, false);
  }
  // pos = Hin @ w3 + b3  (16 rows x 3 dims = 48 serial dots)
  if (tid < 48) {
    int r = tid / 3, d = tid % 3;
    float s = b3[d];
    for (int c = 0; c < IN_DIM; ++c) s += bf2f(Hin[r * ldH + c]) * w3[c * 3 + d];
    pos[r * 3 + d] = s;
  }
  __syncthreads();
  // logits = [Fout, pos] @ rw + rb + 0.5*geo  (T = 1)
  if (tid < 64) {
    int r = tid >> 2, j = tid & 3;
    float dx = pos[r * 3 + 0] - centers[j * 3 + 0];
    float dy = pos[r * 3 + 1] - centers[j * 3 + 1];
    float dz = pos[r * 3 + 2] - centers[j * 3 + 2];
    float geo = -(dx * dx + dy * dy + dz * dz) / (2.0f + 1e-8f);
    float s = rb[j] + 0.5f * geo;
    for (int c = 0; c < 128; ++c) s += bf2f(Fout[r * 128 + c]) * rw[c * 4 + j];
    for (int d = 0; d < 3; ++d)  s += pos[r * 3 + d] * rw[(128 + d) * 4 + j];
    lg[r * 4 + j] = s;
  }
  __syncthreads();
  if (tid < 16) { // softmax over 4 (T = 1)
    float m = lg[tid * 4];
    for (int j = 1; j < 4; ++j) m = fmaxf(m, lg[tid * 4 + j]);
    float e[4], sum = 0.f;
    for (int j = 0; j < 4; ++j) { e[j] = expf(lg[tid * 4 + j] - m); sum += e[j]; }
    float inv = 1.0f / sum;
    for (int j = 0; j < 4; ++j) probs[tid * 4 + j] = e[j] * inv;
  }
  __syncthreads();
}

// --- workspace layout (bf16 elements): transposed, bf16-converted weights ----
#define OFF_PW1   0           // 512  x 2048
#define OFF_PW2   1048576     // 256  x 512
#define OFF_L1F1  1179648     // 128  x 256
#define OFF_L1F2  1212416     // 128  x 128
#define OFF_L2F1  1228800     // 128  x 128
#define OFF_L2F2  1245184     // 128  x 128
#define OFF_L3F1  1261568     // 128  x 128
#define OFF_L3F2  1277952     // 128  x 128
#define OFF_HW1   1294336     // 256  x 160  (K padded 140 -> 160 with zeros)
#define OFF_HW2   1335296     // 64   x 256

// dst[n*Kpad + k] = bf16( k < K ? src[k*N + n] : 0 )
__global__ __launch_bounds__(256)
void transpose_to_bf16(const float* __restrict__ src, __bf16* __restrict__ dst,
                       int K, int N, int Kpad) {
  int idx = blockIdx.x * 256 + threadIdx.x;
  if (idx >= N * Kpad) return;
  int n = idx / Kpad, k = idx - n * Kpad;
  float v = (k < K) ? src[(size_t)k * N + n] : 0.0f;
  dst[idx] = f2bf(v);
}

__global__ __launch_bounds__(256)
void bvh_router_fused(const float* __restrict__ x, const __bf16* __restrict__ wsbf,
                      const float* pb1, const float* pb2,
                      const float* lng, const float* lnb,
                      const float* l1_w3, const float* l1_b3, const float* l1_c,
                      const float* l1_f1b, const float* l1_f2b,
                      const float* l1_rw, const float* l1_rb,
                      const float* l2_w3, const float* l2_b3, const float* l2_c,
                      const float* l2_f1b, const float* l2_f2b,
                      const float* l2_rw, const float* l2_rb,
                      const float* l3_w3, const float* l3_b3, const float* l3_c,
                      const float* l3_f1b, const float* l3_f2b,
                      const float* l3_rw, const float* l3_rb,
                      const float* hb1, const float* hb2,
                      float* __restrict__ out_probs, float* __restrict__ out_ids) {
  // bf16 LDS arena (elements), aliased across phases:
  //   stage 1/2 : Xb [0,8192) (16x512 chunk), H1b [8192,16384) (16x512)
  //   levels    : Hb [0,4096), Tmp [4096,6144), FA [6144,8192), FB [8192,10240)
  //   head      : C16 [10240,12800) (16x160), A1 [12800,16896) (16x256)
  __shared__ __align__(32) __bf16 sb[16896];
  // f32 LDS: pre-LN hidden + final logits
  __shared__ __align__(16) float sf[5120];   // Hf [0,4096), L [4096,5120)
  __shared__ float pos_s[48];
  __shared__ float lg_s[64];
  __shared__ float p1s[64], p2s[64], p3s[64];

  __bf16* Xb  = sb;
  __bf16* H1b = sb + 8192;
  __bf16* Hb  = sb;
  __bf16* Tmp = sb + 4096;
  __bf16* FA  = sb + 6144;
  __bf16* FB  = sb + 8192;
  __bf16* C16 = sb + 10240;
  __bf16* A1  = sb + 12800;
  float*  Hf  = sf;
  float*  L   = sf + 4096;

  const int tid  = threadIdx.x;
  const int wid  = tid >> 5;
  const int row0 = blockIdx.x * 16;
  const float* xrow = x + (size_t)row0 * 2048;

  // Prefetch this block's 128 KB x-tile (global_prefetch_b8).
  for (int off = tid * 64; off < 16 * 2048 * 4; off += 256 * 64)
    __builtin_prefetch((const char*)xrow + off, 0, 1);

  // ---- Stage 1: H1 = GELU(x @ pw1 + pb1), K=2048 in 4 chunks of 512, N=512 --
  v8f acc4[4] = {zero8(), zero8(), zero8(), zero8()};
  for (int kc = 0; kc < 2048; kc += 512) {
    { // stage x chunk into LDS as bf16 (convert once, reuse for 4 N-tiles)
      int r = tid >> 4, c0 = (tid & 15) * 32;
      const float* src = xrow + r * 2048 + kc + c0;
      __bf16* dst = Xb + r * 512 + c0;
#pragma unroll
      for (int q = 0; q < 8; ++q) {
        float4 v = *(const float4*)(src + q * 4);
        dst[q * 4 + 0] = f2bf(v.x); dst[q * 4 + 1] = f2bf(v.y);
        dst[q * 4 + 2] = f2bf(v.z); dst[q * 4 + 3] = f2bf(v.w);
      }
    }
    __syncthreads();
#pragma unroll
    for (int t = 0; t < 4; ++t) {
      int n0 = (wid + t * 8) * 16;
      acc4[t] = wmma_tileB<512>(Xb, 512,
                                wsbf + OFF_PW1 + (size_t)n0 * 2048 + kc, 2048,
                                acc4[t]);
    }
    __syncthreads();
  }
#pragma unroll
  for (int t = 0; t < 4; ++t)
    store_tile_bf(H1b, 512, (wid + t * 8) * 16, pb1, acc4[t], true);
  __syncthreads();

  // ---- Stage 2: Hf = H1 @ pw2 + pb2 (pre-LN, f32), K=512, N=256 -------------
  for (int t = 0; t < 2; ++t) {
    int n0 = (wid + t * 8) * 16;
    v8f acc = wmma_tileB<512>(H1b, 512, wsbf + OFF_PW2 + n0 * 512, 512, zero8());
    store_tile_f32(Hf, 256, n0, pb2, acc);
  }
  __syncthreads();

  // LayerNorm over 256 per row (f32), then convert once to bf16 Hb
  if (tid < 16) {
    float mu = 0.f;
    for (int c = 0; c < 256; ++c) mu += Hf[tid * 256 + c];
    mu *= (1.0f / 256.0f);
    float var = 0.f;
    for (int c = 0; c < 256; ++c) { float d = Hf[tid * 256 + c] - mu; var += d * d; }
    var *= (1.0f / 256.0f);
    float inv = rsqrtf(var + 1e-5f);
    for (int c = 0; c < 256; ++c)
      Hf[tid * 256 + c] = (Hf[tid * 256 + c] - mu) * inv * lng[c] + lnb[c];
  }
  __syncthreads();
  for (int idx = tid; idx < 4096; idx += 256) Hb[idx] = f2bf(Hf[idx]);
  __syncthreads();

  // ---- Three routing levels: f1 -> FA, f2 -> FB, f3 -> FA -------------------
  run_level<256>(Hb, 256, l1_w3, l1_b3, l1_c,
                 wsbf + OFF_L1F1, l1_f1b, wsbf + OFF_L1F2, l1_f2b,
                 l1_rw, l1_rb, Tmp, FA, pos_s, lg_s, p1s);
  run_level<128>(FA, 128, l2_w3, l2_b3, l2_c,
                 wsbf + OFF_L2F1, l2_f1b, wsbf + OFF_L2F2, l2_f2b,
                 l2_rw, l2_rb, Tmp, FB, pos_s, lg_s, p2s);
  run_level<128>(FB, 128, l3_w3, l3_b3, l3_c,
                 wsbf + OFF_L3F1, l3_f1b, wsbf + OFF_L3F2, l3_f2b,
                 l3_rw, l3_rb, Tmp, FA, pos_s, lg_s, p3s);

  // combined = [f3(128), p1(4), p2(4), p3(4), zeros(20)]  (16 x 160, padded)
  for (int idx = tid; idx < 16 * 160; idx += 256) {
    int r = idx / 160, c = idx - (idx / 160) * 160;
    float v = 0.0f;
    if (c < 128)      v = bf2f(FA[r * 128 + c]);
    else if (c < 132) v = p1s[r * 4 + (c - 128)];
    else if (c < 136) v = p2s[r * 4 + (c - 132)];
    else if (c < 140) v = p3s[r * 4 + (c - 136)];
    C16[r * 160 + c] = f2bf(v);
  }
  __syncthreads();

  // ---- Head 1: A1 = GELU(C16 @ hw1 + hb1), K=160 (zero padded), N=256 -------
  for (int t = 0; t < 2; ++t) {
    int n0 = (wid + t * 8) * 16;
    v8f acc = wmma_tileB<160>(C16, 160, wsbf + OFF_HW1 + n0 * 160, 160, zero8());
    store_tile_bf(A1, 256, n0, hb1, acc, true);
  }
  __syncthreads();

  // ---- Head 2: L = A1 @ hw2 + hb2, K=256, N=64 (waves 0..3, uniform) --------
  if (wid < 4) {
    int n0 = wid * 16;
    v8f acc = wmma_tileB<256>(A1, 256, wsbf + OFF_HW2 + n0 * 256, 256, zero8());
    store_tile_f32(L, 64, n0, hb2, acc);
  }
  __syncthreads();

  // ---- Softmax over 64 (T = 1) + argmax (first max) -------------------------
  if (tid < 16) {
    int r = tid;
    float m = L[r * 64]; int bi = 0;
    for (int j = 1; j < 64; ++j) {
      float v = L[r * 64 + j];
      if (v > m) { m = v; bi = j; }
    }
    float sum = 0.f;
    for (int j = 0; j < 64; ++j) {
      float e = expf(L[r * 64 + j] - m);
      L[r * 64 + j] = e;
      sum += e;
    }
    float inv = 1.0f / sum;
    float* op = out_probs + (size_t)(row0 + r) * 64;
    for (int j = 0; j < 64; ++j) op[j] = L[r * 64 + j] * inv;
    out_ids[row0 + r] = (float)bi;
  }
}

static inline void launch_tr(const float* src, __bf16* dst, int K, int N, int Kpad,
                             hipStream_t stream) {
  int elems = N * Kpad;
  transpose_to_bf16<<<(elems + 255) / 256, 256, 0, stream>>>(src, dst, K, N, Kpad);
}

extern "C" void kernel_launch(void* const* d_in, const int* in_sizes, int n_in,
                              void* d_out, int out_size, void* d_ws, size_t ws_size,
                              hipStream_t stream) {
  (void)n_in; (void)out_size; (void)ws_size;
  const float* x    = (const float*)d_in[0];
  const float* pw1  = (const float*)d_in[1];
  const float* pb1  = (const float*)d_in[2];
  const float* pw2  = (const float*)d_in[3];
  const float* pb2  = (const float*)d_in[4];
  const float* lng  = (const float*)d_in[5];
  const float* lnb  = (const float*)d_in[6];
  const float* l1_w3  = (const float*)d_in[7];
  const float* l1_b3  = (const float*)d_in[8];
  const float* l1_c   = (const float*)d_in[9];
  const float* l1_f1w = (const float*)d_in[10];
  const float* l1_f1b = (const float*)d_in[11];
  const float* l1_f2w = (const float*)d_in[12];
  const float* l1_f2b = (const float*)d_in[13];
  const float* l1_rw  = (const float*)d_in[14];
  const float* l1_rb  = (const float*)d_in[15];
  const float* l2_w3  = (const float*)d_in[16];
  const float* l2_b3  = (const float*)d_in[17];
  const float* l2_c   = (const float*)d_in[18];
  const float* l2_f1w = (const float*)d_in[19];
  const float* l2_f1b = (const float*)d_in[20];
  const float* l2_f2w = (const float*)d_in[21];
  const float* l2_f2b = (const float*)d_in[22];
  const float* l2_rw  = (const float*)d_in[23];
  const float* l2_rb  = (const float*)d_in[24];
  const float* l3_w3  = (const float*)d_in[25];
  const float* l3_b3  = (const float*)d_in[26];
  const float* l3_c   = (const float*)d_in[27];
  const float* l3_f1w = (const float*)d_in[28];
  const float* l3_f1b = (const float*)d_in[29];
  const float* l3_f2w = (const float*)d_in[30];
  const float* l3_f2b = (const float*)d_in[31];
  const float* l3_rw  = (const float*)d_in[32];
  const float* l3_rb  = (const float*)d_in[33];
  const float* hw1  = (const float*)d_in[34];
  const float* hb1  = (const float*)d_in[35];
  const float* hw2  = (const float*)d_in[36];
  const float* hb2  = (const float*)d_in[37];

  __bf16* wsbf = (__bf16*)d_ws;

  // Weight prep: transpose + bf16 convert into workspace (all L2-resident).
  launch_tr(pw1,    wsbf + OFF_PW1,  2048, 512, 2048, stream);
  launch_tr(pw2,    wsbf + OFF_PW2,  512,  256, 512,  stream);
  launch_tr(l1_f1w, wsbf + OFF_L1F1, 256,  128, 256,  stream);
  launch_tr(l1_f2w, wsbf + OFF_L1F2, 128,  128, 128,  stream);
  launch_tr(l2_f1w, wsbf + OFF_L2F1, 128,  128, 128,  stream);
  launch_tr(l2_f2w, wsbf + OFF_L2F2, 128,  128, 128,  stream);
  launch_tr(l3_f1w, wsbf + OFF_L3F1, 128,  128, 128,  stream);
  launch_tr(l3_f2w, wsbf + OFF_L3F2, 128,  128, 128,  stream);
  launch_tr(hw1,    wsbf + OFF_HW1,  140,  256, 160,  stream);
  launch_tr(hw2,    wsbf + OFF_HW2,  256,  64,  256,  stream);

  const int Brows = in_sizes[0] / 2048;           // 32768
  float* out_probs = (float*)d_out;
  float* out_ids   = out_probs + (size_t)Brows * 64;

  dim3 grid(Brows / 16), block(256);
  bvh_router_fused<<<grid, block, 0, stream>>>(
      x, wsbf, pb1, pb2, lng, lnb,
      l1_w3, l1_b3, l1_c, l1_f1b, l1_f2b, l1_rw, l1_rb,
      l2_w3, l2_b3, l2_c, l2_f1b, l2_f2b, l2_rw, l2_rb,
      l3_w3, l3_b3, l3_c, l3_f1b, l3_f2b, l3_rw, l3_rb,
      hb1, hb2, out_probs, out_ids);
}
